// NearestNeighborTokenizer_884763263649
// MI455X (gfx1250) — matrix operand order repaired and checked by reading, not compiled, not run on hardware
//
#include <hip/hip_runtime.h>
#include <hip/hip_bf16.h>
#include <stdint.h>

// CDNA5 WMMA vector types
typedef __attribute__((ext_vector_type(16))) __bf16 v16bf;
typedef __attribute__((ext_vector_type(8)))  float  v8f;
typedef __attribute__((ext_vector_type(4)))  unsigned int u32x4;
typedef __attribute__((ext_vector_type(8)))  int  i32x8;
typedef __attribute__((ext_vector_type(4)))  int  i32x4;

#define TOK      8192          // 4 * 2048 tokens
#define DDIM     512           // feature dim
#define NCODES   16384         // codebook size
#define MT       64            // tokens per workgroup
#define CCHUNK   2048          // codes per workgroup (grid.y split)
#define NBLK     64            // codes per LDS block (fallback path)
#define NBLK2    32            // codes per LDS block (TDM double-buffered path)
#define NCHUNKS  (NCODES / CCHUNK)   // 8
#define ASTRIDE  520           // padded LDS row stride (bf16 elems), bank-conflict free
#define THRESH   900.0f

// ---- LDS byte-offset carve for the TDM (pre-converted) phase-1 kernel ----
#define L_AHI    0u
#define L_ALO    66560u                 // 64*520*2
#define L_BHI    133120u                // two buffers of 32*520*2 = 33280 each
#define L_BLO    199680u
#define L_BBUF   33280u
#define L_RED    266240u
#define L_TOTAL  282624u                // + 2*8192 reduction

#if defined(__has_builtin)
#if __has_builtin(__builtin_amdgcn_tensor_load_to_lds)
#define HAVE_TDM 1
#endif
#endif

union FragBF { uint4 q[2]; v16bf v; };

__device__ __forceinline__ unsigned short f32_to_bf16(float f) {
  union { float f; unsigned u; } v; v.f = f;
  unsigned u = v.u;
  u += 0x7FFFu + ((u >> 16) & 1u);   // round-to-nearest-even
  return (unsigned short)(u >> 16);
}
__device__ __forceinline__ float bf16_to_f32(unsigned short h) {
  union { unsigned u; float f; } v; v.u = ((unsigned)h) << 16;
  return v.f;
}

#ifdef HAVE_TDM
// TDM 2D tile load: rows x 512 bf16 elements, row-major (stride 512 elems),
// LDS destination padded by 4 DWORDs every 256 DWORDs -> 520-ushort row stride.
// D# built per CDNA5 ISA ch8 (group0: count/lds/global/type, group1: dims).
__device__ __forceinline__ void tdm_load_2d(unsigned ldsAddr, const void* gaddr,
                                            unsigned rows) {
  const unsigned long long ga = (unsigned long long)(uintptr_t)gaddr;
  u32x4 g0;
  g0[0] = 1u;                                               // count=1, user mode
  g0[1] = ldsAddr;                                          // lds_addr (bytes)
  g0[2] = (unsigned)(ga & 0xFFFFFFFFu);                     // global_addr[31:0]
  g0[3] = (unsigned)((ga >> 32) & 0x01FFFFFFu) | (2u << 30); // addr[56:32] | type=2
  i32x8 g1;
  g1[0] = (int)((1u << 16)        // data_size = 2 bytes
              | (1u << 20)        // pad_enable
              | (7u << 22)        // pad_interval: 256 DWORDs (one 1024B row)
              | (3u << 25));      // pad_amount: 4 DWORDs (16B) -> stride 520 ushorts
  g1[1] = (int)(512u << 16);      // tensor_dim0 = 512 (bits 79:48)
  g1[2] = (int)(rows << 16);      // tensor_dim1 = rows (bits 111:80)
  g1[3] = (int)(512u << 16);      // tile_dim0 = 512 (bits 127:112)
  g1[4] = (int)rows;              // tile_dim1 = rows; tile_dim2 = 0
  g1[5] = 512;                    // tensor_dim0_stride = 512 elems (bits 207:160)
  g1[6] = 0;
  g1[7] = 0;
  i32x4 z4 = {0, 0, 0, 0};
#if __clang_major__ >= 23
  i32x8 z8 = {0, 0, 0, 0, 0, 0, 0, 0};
  __builtin_amdgcn_tensor_load_to_lds(g0, g1, z4, z4, z8, 0);
#else
  __builtin_amdgcn_tensor_load_to_lds(g0, g1, z4, z4, 0);
#endif
}
#endif

// ============================================================================
// Pre-pass: split fp32 rows into bf16 hi/lo planes + exact fp32 row norms.
// ============================================================================
__global__ void __launch_bounds__(256)
nn_convert(const float* __restrict__ src, unsigned short* __restrict__ dHi,
           unsigned short* __restrict__ dLo, float* __restrict__ norms, int nrows)
{
  const int lane = threadIdx.x & 31;
  const int w    = threadIdx.x >> 5;
  const int r    = blockIdx.x * 8 + w;
  if (r >= nrows) return;
  const float* s = src + (size_t)r * DDIM + lane * 16;
  __align__(16) unsigned short hbuf[16], lbuf[16];
  float sq = 0.f;
#pragma unroll
  for (int c4 = 0; c4 < 4; ++c4) {
    float4 v4 = ((const float4*)s)[c4];
    float e[4] = {v4.x, v4.y, v4.z, v4.w};
#pragma unroll
    for (int k = 0; k < 4; ++k) {
      float f = e[k];
      sq += f * f;
      unsigned short h = f32_to_bf16(f);
      hbuf[c4*4 + k] = h;
      lbuf[c4*4 + k] = f32_to_bf16(f - bf16_to_f32(h));
    }
  }
  unsigned short* oH = dHi + (size_t)r * DDIM + lane * 16;
  unsigned short* oL = dLo + (size_t)r * DDIM + lane * 16;
  ((uint4*)oH)[0] = ((const uint4*)hbuf)[0];
  ((uint4*)oH)[1] = ((const uint4*)hbuf)[1];
  ((uint4*)oL)[0] = ((const uint4*)lbuf)[0];
  ((uint4*)oL)[1] = ((const uint4*)lbuf)[1];
#pragma unroll
  for (int off = 16; off > 0; off >>= 1) sq += __shfl_xor(sq, off, 32);
  if (lane == 0) norms[r] = sq;
}

// ============================================================================
// Phase 1 (TDM path): pre-converted bf16 planes; double-buffered code blocks
// staged by the Tensor Data Mover (TENSORcnt); fused WMMA + argmin.
// ============================================================================
__global__ void __launch_bounds__(256, 1)
nn_phase1_pre(const unsigned short* __restrict__ xHi, const unsigned short* __restrict__ xLo,
              const unsigned short* __restrict__ cHi, const unsigned short* __restrict__ cLo,
              const float* __restrict__ x2g, const float* __restrict__ c2g,
              float* __restrict__ wsD, int* __restrict__ wsI)
{
  extern __shared__ __align__(16) char smem_raw[];
  unsigned short* aHi  = (unsigned short*)(smem_raw + L_AHI);
  unsigned short* aLo  = (unsigned short*)(smem_raw + L_ALO);
  unsigned short* bHiB = (unsigned short*)(smem_raw + L_BHI);
  unsigned short* bLoB = (unsigned short*)(smem_raw + L_BLO);
  float* redD = (float*)(smem_raw + L_RED);
  int*   redI = (int*)(smem_raw + L_RED + 8192);

  const unsigned ldsDyn = __builtin_amdgcn_groupstaticsize();  // dynamic-LDS base

  const int tid   = threadIdx.x;
  const int lane  = tid & 31;
  const int w     = tid >> 5;          // wave 0..7
  const int mlane = lane & 15;
  const int hi16  = lane >> 4;
  const int rq    = w & 3;             // row group: rows rq*16 .. rq*16+15
  const int tg    = w >> 2;            // tile group: N-tile tg (of 2)
  const int rowbase   = rq * 16;
  const int tokBase   = blockIdx.x * MT;
  const int chunkBase = blockIdx.y * CCHUNK;

  // ---- stage token tile: raw b128 copies of pre-converted planes ----
#pragma unroll
  for (int i = 0; i < 16; ++i) {
    const int c   = i * 256 + tid;
    const int r   = c >> 6;
    const int col = (c & 63) * 8;      // ushort column
    *(uint4*)(aHi + r * ASTRIDE + col) = *(const uint4*)(xHi + (size_t)(tokBase + r) * DDIM + col);
    *(uint4*)(aLo + r * ASTRIDE + col) = *(const uint4*)(xLo + (size_t)(tokBase + r) * DDIM + col);
  }

  float x2j[8];
#pragma unroll
  for (int j = 0; j < 8; ++j) x2j[j] = x2g[tokBase + rowbase + j + hi16 * 8];

  float bestD[8];
  int   bestI[8];
#pragma unroll
  for (int j = 0; j < 8; ++j) { bestD[j] = 3.4e38f; bestI[j] = -1; }

  // stage one 32-code block into LDS buffer `buf`
  auto stage = [&](int nb, int buf) {
    const unsigned ldsHi = ldsDyn + L_BHI + (unsigned)buf * L_BBUF;
    const unsigned ldsLo = ldsDyn + L_BLO + (unsigned)buf * L_BBUF;
    const unsigned short* gH = cHi + (size_t)(chunkBase + nb * NBLK2) * DDIM;
    const unsigned short* gL = cLo + (size_t)(chunkBase + nb * NBLK2) * DDIM;
#ifdef HAVE_TDM
    if (w == 0) {                       // TDM issued once; EXEC ignored by TDM
      tdm_load_2d(ldsHi, gH, NBLK2);
      tdm_load_2d(ldsLo, gL, NBLK2);
    }
#else
    const unsigned long long gHa = (unsigned long long)(uintptr_t)gH;
    const unsigned long long gLa = (unsigned long long)(uintptr_t)gL;
#pragma unroll
    for (int i = 0; i < 8; ++i) {
      const int c   = i * 256 + tid;
      const int r   = c >> 6;
      const int col = (c & 63) * 8;
      const unsigned ldsOff = (unsigned)(r * ASTRIDE + col) * 2u;
      const unsigned gOff   = (unsigned)(r * DDIM + col) * 2u;
      asm volatile("global_load_async_to_lds_b128 %0, %1, %2"
                   :: "v"(ldsHi + ldsOff), "v"(gOff), "s"(gHa) : "memory");
      asm volatile("global_load_async_to_lds_b128 %0, %1, %2"
                   :: "v"(ldsLo + ldsOff), "v"(gOff), "s"(gLa) : "memory");
    }
#endif
  };
  auto stage_wait = [&]() {
#ifdef HAVE_TDM
    __builtin_amdgcn_s_wait_tensorcnt(0);   // no-op for waves with TENSORcnt==0
#else
    asm volatile("s_wait_asynccnt 0x0" ::: "memory");
#endif
  };

  stage(0, 0);
  stage_wait();
  __syncthreads();

  const int NBLOCKS = CCHUNK / NBLK2;  // 64
  for (int nb = 0; nb < NBLOCKS; ++nb) {
    const int buf = nb & 1;
    if (nb + 1 < NBLOCKS) stage(nb + 1, buf ^ 1);   // overlap with compute below

    const unsigned short* bHi = bHiB + (size_t)buf * (L_BBUF / 2);
    const unsigned short* bLo = bLoB + (size_t)buf * (L_BBUF / 2);

    const int   id  = chunkBase + nb * NBLK2 + tg * 16 + mlane;
    const float c2v = c2g[id];

    // two independent accumulator chains (even/odd K-steps) for WMMA ILP
    v8f accA = {0.f,0.f,0.f,0.f,0.f,0.f,0.f,0.f};
    v8f accB = accA;

#pragma unroll 4
    for (int ks2 = 0; ks2 < 8; ++ks2) {
      const int ksA = ks2 * 2, ksB = ks2 * 2 + 1;
      const int aoffA = (rowbase + mlane) * ASTRIDE + ksA * 32 + hi16 * 16;
      const int aoffB = (rowbase + mlane) * ASTRIDE + ksB * 32 + hi16 * 16;
      const int boffA = (tg * 16 + mlane) * ASTRIDE + ksA * 32 + hi16 * 16;
      const int boffB = (tg * 16 + mlane) * ASTRIDE + ksB * 32 + hi16 * 16;
      FragBF ahA, alA, bhA, blA, ahB, alB, bhB, blB;
      ahA.q[0] = *(const uint4*)(aHi + aoffA);  ahA.q[1] = *(const uint4*)(aHi + aoffA + 8);
      alA.q[0] = *(const uint4*)(aLo + aoffA);  alA.q[1] = *(const uint4*)(aLo + aoffA + 8);
      bhA.q[0] = *(const uint4*)(bHi + boffA);  bhA.q[1] = *(const uint4*)(bHi + boffA + 8);
      blA.q[0] = *(const uint4*)(bLo + boffA);  blA.q[1] = *(const uint4*)(bLo + boffA + 8);
      ahB.q[0] = *(const uint4*)(aHi + aoffB);  ahB.q[1] = *(const uint4*)(aHi + aoffB + 8);
      alB.q[0] = *(const uint4*)(aLo + aoffB);  alB.q[1] = *(const uint4*)(aLo + aoffB + 8);
      bhB.q[0] = *(const uint4*)(bHi + boffB);  bhB.q[1] = *(const uint4*)(bHi + boffB + 8);
      blB.q[0] = *(const uint4*)(bLo + boffB);  blB.q[1] = *(const uint4*)(bLo + boffB + 8);
      // split-precision fp32 emulation: xh*ch + xl*ch + xh*cl (interleaved chains)
      accA = __builtin_amdgcn_wmma_f32_16x16x32_bf16(false, ahA.v, false, bhA.v, (short)0, accA, false, false);
      accB = __builtin_amdgcn_wmma_f32_16x16x32_bf16(false, ahB.v, false, bhB.v, (short)0, accB, false, false);
      accA = __builtin_amdgcn_wmma_f32_16x16x32_bf16(false, alA.v, false, bhA.v, (short)0, accA, false, false);
      accB = __builtin_amdgcn_wmma_f32_16x16x32_bf16(false, alB.v, false, bhB.v, (short)0, accB, false, false);
      accA = __builtin_amdgcn_wmma_f32_16x16x32_bf16(false, ahA.v, false, blA.v, (short)0, accA, false, false);
      accB = __builtin_amdgcn_wmma_f32_16x16x32_bf16(false, ahB.v, false, blB.v, (short)0, accB, false, false);
    }
    const v8f acc = accA + accB;

#pragma unroll
    for (int j = 0; j < 8; ++j) {
      float d = fmaxf(x2j[j] + c2v - 2.f * acc[j], 0.f);
      if (d < bestD[j] || (d == bestD[j] && id < bestI[j])) { bestD[j] = d; bestI[j] = id; }
    }

    stage_wait();
    __syncthreads();
  }

  // ---- cross-lane / cross-wave argmin reduction via LDS ----
#pragma unroll
  for (int j = 0; j < 8; ++j) {
    const int idx = ((rq * 8 + j) * 2 + tg) * 32 + lane;
    redD[idx] = bestD[j];
    redI[idx] = bestI[j];
  }
  __syncthreads();
  if (w < 4 && lane < 16) {
    const int j = lane & 7, h = lane >> 3;   // row in tile = w*16 + lane
    float bd = 3.4e38f; int bi = -1;
#pragma unroll
    for (int tg2 = 0; tg2 < 2; ++tg2) {
      const int base = ((w * 8 + j) * 2 + tg2) * 32 + h * 16;
      for (int s = 0; s < 16; ++s) {
        const float d  = redD[base + s];
        const int   iv = redI[base + s];
        if (d < bd || (d == bd && iv < bi)) { bd = d; bi = iv; }
      }
    }
    const int tok = tokBase + w * 16 + lane;
    wsD[(size_t)blockIdx.y * TOK + tok] = bd;
    wsI[(size_t)blockIdx.y * TOK + tok] = bi;
  }
}

// ============================================================================
// Phase 1 (fallback, small workspace): self-converting version.
// ============================================================================
__global__ void __launch_bounds__(256, 1)
nn_phase1(const float* __restrict__ x, const float* __restrict__ codes,
          float* __restrict__ wsD, int* __restrict__ wsI)
{
  extern __shared__ __align__(16) char smem_raw[];
  unsigned short* aHi = (unsigned short*)smem_raw;        // [MT][ASTRIDE]
  unsigned short* aLo = aHi + MT * ASTRIDE;
  unsigned short* bHi = aLo + MT * ASTRIDE;               // [NBLK][ASTRIDE]
  unsigned short* bLo = bHi + NBLK * ASTRIDE;
  float* x2s  = (float*)(bLo + NBLK * ASTRIDE);
  float* c2s  = x2s + MT;
  float* redD = c2s + NBLK;
  int*   redI = (int*)(redD + 4 * 8 * 2 * 32);

  const int tid   = threadIdx.x;
  const int lane  = tid & 31;
  const int w     = tid >> 5;
  const int mlane = lane & 15;
  const int hi16  = lane >> 4;
  const int rq    = w & 3;
  const int tg    = w >> 2;
  const int rowbase   = rq * 16;
  const int tokBase   = blockIdx.x * MT;
  const int chunkBase = blockIdx.y * CCHUNK;

  for (int i = 0; i < 8; ++i) {
    const int r = w * 8 + i;
    const float* src = x + (size_t)(tokBase + r) * DDIM + lane * 16;
    unsigned short* dH = aHi + r * ASTRIDE + lane * 16;
    unsigned short* dL = aLo + r * ASTRIDE + lane * 16;
    float sq = 0.f;
#pragma unroll
    for (int c4 = 0; c4 < 4; ++c4) {
      float4 v4 = ((const float4*)src)[c4];
      float e[4] = {v4.x, v4.y, v4.z, v4.w};
#pragma unroll
      for (int k = 0; k < 4; ++k) {
        float f = e[k];
        sq += f * f;
        unsigned short h = f32_to_bf16(f);
        dH[c4*4 + k] = h;
        dL[c4*4 + k] = f32_to_bf16(f - bf16_to_f32(h));
      }
    }
#pragma unroll
    for (int off = 16; off > 0; off >>= 1) sq += __shfl_xor(sq, off, 32);
    if (lane == 0) x2s[r] = sq;
  }
  __syncthreads();

  float x2j[8];
#pragma unroll
  for (int j = 0; j < 8; ++j) x2j[j] = x2s[rowbase + j + hi16 * 8];

  float bestD[8];
  int   bestI[8];
#pragma unroll
  for (int j = 0; j < 8; ++j) { bestD[j] = 3.4e38f; bestI[j] = -1; }

  for (int nb = 0; nb < CCHUNK / NBLK; ++nb) {
    for (int i = 0; i < 8; ++i) {
      const int r = w * 8 + i;
      const float* src = codes + (size_t)(chunkBase + nb * NBLK + r) * DDIM + lane * 16;
      unsigned short* dH = bHi + r * ASTRIDE + lane * 16;
      unsigned short* dL = bLo + r * ASTRIDE + lane * 16;
      float sq = 0.f;
#pragma unroll
      for (int c4 = 0; c4 < 4; ++c4) {
        float4 v4 = ((const float4*)src)[c4];
        float e[4] = {v4.x, v4.y, v4.z, v4.w};
#pragma unroll
        for (int k = 0; k < 4; ++k) {
          float f = e[k];
          sq += f * f;
          unsigned short h = f32_to_bf16(f);
          dH[c4*4 + k] = h;
          dL[c4*4 + k] = f32_to_bf16(f - bf16_to_f32(h));
        }
      }
#pragma unroll
      for (int off = 16; off > 0; off >>= 1) sq += __shfl_xor(sq, off, 32);
      if (lane == 0) c2s[r] = sq;
    }
    __syncthreads();

    if (nb + 1 < CCHUNK / NBLK) {
      const float* pf = codes + (size_t)(chunkBase + (nb + 1) * NBLK + (tid >> 2)) * DDIM
                        + (tid & 3) * 128;
      __builtin_prefetch(pf, 0, 1);
    }

    v8f acc0 = {0.f,0.f,0.f,0.f,0.f,0.f,0.f,0.f};
    v8f acc1 = acc0;

#pragma unroll 4
    for (int ks = 0; ks < 16; ++ks) {
      const int aoff = (rowbase + mlane) * ASTRIDE + ks * 32 + hi16 * 16;
      FragBF ah, al;
      ah.q[0] = *(const uint4*)(aHi + aoff);
      ah.q[1] = *(const uint4*)(aHi + aoff + 8);
      al.q[0] = *(const uint4*)(aLo + aoff);
      al.q[1] = *(const uint4*)(aLo + aoff + 8);

      const int b0 = ((tg * 2 + 0) * 16 + mlane) * ASTRIDE + ks * 32 + hi16 * 16;
      const int b1 = ((tg * 2 + 1) * 16 + mlane) * ASTRIDE + ks * 32 + hi16 * 16;
      FragBF bh0, bl0, bh1, bl1;
      bh0.q[0] = *(const uint4*)(bHi + b0);  bh0.q[1] = *(const uint4*)(bHi + b0 + 8);
      bl0.q[0] = *(const uint4*)(bLo + b0);  bl0.q[1] = *(const uint4*)(bLo + b0 + 8);
      bh1.q[0] = *(const uint4*)(bHi + b1);  bh1.q[1] = *(const uint4*)(bHi + b1 + 8);
      bl1.q[0] = *(const uint4*)(bLo + b1);  bl1.q[1] = *(const uint4*)(bLo + b1 + 8);

      acc0 = __builtin_amdgcn_wmma_f32_16x16x32_bf16(false, ah.v, false, bh0.v, (short)0, acc0, false, false);
      acc1 = __builtin_amdgcn_wmma_f32_16x16x32_bf16(false, ah.v, false, bh1.v, (short)0, acc1, false, false);
      acc0 = __builtin_amdgcn_wmma_f32_16x16x32_bf16(false, al.v, false, bh0.v, (short)0, acc0, false, false);
      acc1 = __builtin_amdgcn_wmma_f32_16x16x32_bf16(false, al.v, false, bh1.v, (short)0, acc1, false, false);
      acc0 = __builtin_amdgcn_wmma_f32_16x16x32_bf16(false, ah.v, false, bl0.v, (short)0, acc0, false, false);
      acc1 = __builtin_amdgcn_wmma_f32_16x16x32_bf16(false, ah.v, false, bl1.v, (short)0, acc1, false, false);
    }

    const int codeBase = chunkBase + nb * NBLK;
#pragma unroll
    for (int tt = 0; tt < 2; ++tt) {
      const int t = tg * 2 + tt;
      const float c2v = c2s[t * 16 + mlane];
      const int   id  = codeBase + t * 16 + mlane;
      const v8f acc = tt ? acc1 : acc0;
#pragma unroll
      for (int j = 0; j < 8; ++j) {
        float d = fmaxf(x2j[j] + c2v - 2.f * acc[j], 0.f);
        if (d < bestD[j] || (d == bestD[j] && id < bestI[j])) { bestD[j] = d; bestI[j] = id; }
      }
    }
    __syncthreads();
  }

#pragma unroll
  for (int j = 0; j < 8; ++j) {
    const int idx = ((rq * 8 + j) * 2 + tg) * 32 + lane;
    redD[idx] = bestD[j];
    redI[idx] = bestI[j];
  }
  __syncthreads();
  if (w < 4 && lane < 16) {
    const int j = lane & 7, h = lane >> 3;
    float bd = 3.4e38f; int bi = -1;
#pragma unroll
    for (int tg2 = 0; tg2 < 2; ++tg2) {
      const int base = ((w * 8 + j) * 2 + tg2) * 32 + h * 16;
      for (int s = 0; s < 16; ++s) {
        const float d  = redD[base + s];
        const int   iv = redI[base + s];
        if (d < bd || (d == bd && iv < bi)) { bd = d; bi = iv; }
      }
    }
    const int tok = tokBase + w * 16 + lane;
    wsD[(size_t)blockIdx.y * TOK + tok] = bd;
    wsI[(size_t)blockIdx.y * TOK + tok] = bi;
  }
}

// ============================================================================
// Phase 2: reduce chunk partials per token, apply threshold / sentinel.
// ============================================================================
__global__ void nn_phase2(const float* __restrict__ wsD, const int* __restrict__ wsI,
                          int* __restrict__ out)
{
  const int tok = blockIdx.x * blockDim.x + threadIdx.x;
  if (tok >= TOK) return;
  float bd = 3.4e38f; int bi = -1;
#pragma unroll
  for (int c = 0; c < NCHUNKS; ++c) {
    const float d  = wsD[(size_t)c * TOK + tok];
    const int   iv = wsI[(size_t)c * TOK + tok];
    if (d < bd || (d == bd && iv < bi)) { bd = d; bi = iv; }
  }
  out[tok] = (bd <= THRESH) ? bi : -1;
}

// ============================================================================
extern "C" void kernel_launch(void* const* d_in, const int* in_sizes, int n_in,
                              void* d_out, int out_size, void* d_ws, size_t ws_size,
                              hipStream_t stream) {
  (void)in_sizes; (void)n_in; (void)out_size;
  const float* x     = (const float*)d_in[0];   // [4,2048,512] f32
  const float* codes = (const float*)d_in[1];   // [16384,512] f32
  int* out = (int*)d_out;                       // [8192] int32 ids
  char* ws = (char*)d_ws;

  // big-workspace layout (byte offsets)
  const size_t o_xHi = 0;
  const size_t o_xLo = o_xHi + (size_t)TOK * DDIM * 2;
  const size_t o_cHi = o_xLo + (size_t)TOK * DDIM * 2;
  const size_t o_cLo = o_cHi + (size_t)NCODES * DDIM * 2;
  const size_t o_x2  = o_cLo + (size_t)NCODES * DDIM * 2;
  const size_t o_c2  = o_x2 + (size_t)TOK * 4;
  const size_t o_wsD = o_c2 + (size_t)NCODES * 4;
  const size_t o_wsI = o_wsD + (size_t)NCHUNKS * TOK * 4;
  const size_t need  = o_wsI + (size_t)NCHUNKS * TOK * 4;       // ~51 MB

  dim3 g1(TOK / MT, NCODES / CCHUNK);   // 128 x 8 workgroups

  if (ws_size >= need) {
    unsigned short* xHi = (unsigned short*)(ws + o_xHi);
    unsigned short* xLo = (unsigned short*)(ws + o_xLo);
    unsigned short* cHi = (unsigned short*)(ws + o_cHi);
    unsigned short* cLo = (unsigned short*)(ws + o_cLo);
    float* x2g = (float*)(ws + o_x2);
    float* c2g = (float*)(ws + o_c2);
    float* wsD = (float*)(ws + o_wsD);
    int*   wsI = (int*)(ws + o_wsI);

    nn_convert<<<TOK / 8, 256, 0, stream>>>(x, xHi, xLo, x2g, TOK);
    nn_convert<<<NCODES / 8, 256, 0, stream>>>(codes, cHi, cLo, c2g, NCODES);

    const size_t smem = L_TOTAL;   // 282,624 B < 320 KB WGP LDS
    (void)hipFuncSetAttribute((const void*)nn_phase1_pre,
                              hipFuncAttributeMaxDynamicSharedMemorySize, (int)smem);
    nn_phase1_pre<<<g1, 256, smem, stream>>>(xHi, xLo, cHi, cLo, x2g, c2g, wsD, wsI);
    nn_phase2<<<(TOK + 255) / 256, 256, 0, stream>>>(wsD, wsI, out);
  } else {
    float* wsD = (float*)ws;
    int*   wsI = (int*)(ws + sizeof(float) * (size_t)NCHUNKS * TOK);
    const size_t smem = (size_t)(2 * MT + 2 * NBLK) * ASTRIDE * sizeof(unsigned short)
                      + (MT + NBLK) * sizeof(float)
                      + (size_t)2 * (4 * 8 * 2 * 32) * sizeof(int);
    (void)hipFuncSetAttribute((const void*)nn_phase1,
                              hipFuncAttributeMaxDynamicSharedMemorySize, (int)smem);
    nn_phase1<<<g1, 256, smem, stream>>>(x, codes, wsD, wsI);
    nn_phase2<<<(TOK + 255) / 256, 256, 0, stream>>>(wsD, wsI, out);
  }
}
